// NVFP4Llama4Experts_16655883173977
// MI455X (gfx1250) — compile-verified, matrix operand order checked by reading, not run on hardware
//
#include <hip/hip_runtime.h>

// ---------------------------------------------------------------------------
// NVFP4 Llama4 experts (gfx1250 / CDNA5, wave32, V_WMMA_F32_16X16X32_F16)
//   gemm1: gu = x @ dequant(Wgu), inter = up * silu(gate)  -> f16 workspace
//   gemm2: out = inter @ dequant(Wd)                        -> fp32
// Software-pipelined: double-buffered LDS in WMMA fragment layout, register
// prefetch of next K-tile, bit-math fp4 decode (value*2^-14 trick), batched
// ds_store_b128 staging, one block-scale load per thread per K-tile.
// ---------------------------------------------------------------------------

typedef _Float16 v16h __attribute__((ext_vector_type(16)));
typedef float    v8f  __attribute__((ext_vector_type(8)));
typedef _Float16 h2   __attribute__((ext_vector_type(2)));

#define H_DIM 2048
#define I_DIM 4096
#define T_DIM 1024
#define E_NUM 8

// Two e2m1 nibbles (lo nibble = even k) -> two fp16 values scaled by 2^-14,
// packed in one u32.  Placing e2m1's emm at fp16 bits [11:9] gives exactly
// decode(c) * 2^-14 for all 16 codes (subnormals included); 2^14 is folded
// into the block scale.
__device__ __forceinline__ unsigned fp4_decode14(unsigned b) {
    return ((b & 0x07u) << 9)  | ((b & 0x08u) << 12) |
           ((b & 0x70u) << 21) | ((b & 0x80u) << 24);
}
__device__ __forceinline__ unsigned mul2h(unsigned w, _Float16 sh) {
    union { unsigned u; h2 v; } cv; cv.u = w;
    cv.v = cv.v * (h2){sh, sh};                 // v_pk_mul_f16
    return cv.u;
}
__device__ __forceinline__ unsigned packf2(float a, float b) {
    union { _Float16 h[2]; unsigned u; } x;
    x.h[0] = (_Float16)a; x.h[1] = (_Float16)b; return x.u;
}
// u32 offset inside one 16x32-f16 fragment region (256 u32) for row/col rc,
// k-group g (g covers k = 8g..8g+7): lane = (rc&15)+16*(g&1), j0 = (g>>1)*8.
__device__ __forceinline__ int frag_off_u32(int rc, int g) {
    return ((rc & 15) + ((g & 1) << 4)) * 8 + ((g >> 1) & 1) * 4;
}

// ---------------------------------------------------------------------------
// GEMM1 + SiLU.  Tile 128(M) x 64(N) over BOTH gate and up halves.
// Waves 4x2: wave -> mfrags {2m,2m+1}, nfrags {2n,2n+1} per half -> 8 wmma,
// fed by 2 A + 4 B fragment loads per K-tile.
// ---------------------------------------------------------------------------
__global__ __launch_bounds__(256)
void gemm1_silu_kernel(const float* __restrict__ xA,    // [E,T,H]
                       const int*   __restrict__ wpkA,  // [E,H/2,2I]
                       const float* __restrict__ wscA,  // [E,H/16,2I]
                       const float* __restrict__ gsp,
                       _Float16*    __restrict__ interA)// [E,T,I]
{
    const int e = blockIdx.z;
    const float* x    = xA    + (size_t)e * T_DIM * H_DIM;
    const int*   wpk  = wpkA  + (size_t)e * (H_DIM / 2) * (2 * I_DIM);
    const float* wsc  = wscA  + (size_t)e * (H_DIM / 16) * (2 * I_DIM);
    _Float16*    inter= interA+ (size_t)e * T_DIM * I_DIM;

    const int tid = threadIdx.x, lane = tid & 31, wid = tid >> 5;

    __shared__ __align__(32) unsigned aU[2][8 * 256];      // 128x32 halves x2
    __shared__ __align__(32) unsigned bU[2][2][4 * 256];   // 2 halves x 64x32

    const float gs = gsp[0];
    const int m0 = blockIdx.y * 128;
    const int n0 = blockIdx.x * 64;

    // A writer: row ar, k-groups {ag0, ag0+1} (16 consecutive k)
    const int ar  = tid >> 1;
    const int ag0 = (tid & 1) * 2;
    const float* aP = x + (size_t)(m0 + ar) * H_DIM + 8 * ag0;

    // B writer: one column (64 gate + 64 up), k-groups {bg0, bg0+1}
    const int bc    = tid & 127;
    const int bhalf = bc >> 6;
    const int bcc   = bc & 63;
    const int bsel  = tid >> 7;                 // 0: k 0..15, 1: k 16..31
    const int bg0   = bsel * 2;
    const int ncolG = n0 + bcc + bhalf * I_DIM;

    float4   apf0, apf1, apf2, apf3;
    unsigned bpf[8];
    float    spf;

    auto prefetch = [&](int k0) {
        const float* ap = aP + k0;
        apf0 = *(const float4*)(ap + 0);
        apf1 = *(const float4*)(ap + 4);
        apf2 = *(const float4*)(ap + 8);
        apf3 = *(const float4*)(ap + 12);
        const int kp0 = (k0 >> 1) + 8 * bsel;
#pragma unroll
        for (int u = 0; u < 8; ++u)
            bpf[u] = (unsigned)wpk[(size_t)(kp0 + u) * (2 * I_DIM) + ncolG];
        spf = wsc[(size_t)((k0 >> 4) + bsel) * (2 * I_DIM) + ncolG];
    };

    const int mrow = wid >> 1, nsel = wid & 1;

    v8f acc[2][2][2];
#pragma unroll
    for (int h = 0; h < 2; ++h)
#pragma unroll
        for (int mm = 0; mm < 2; ++mm)
#pragma unroll
            for (int nf = 0; nf < 2; ++nf) acc[h][mm][nf] = v8f{};

    prefetch(0);
    const int nIt = H_DIM / 32;
    for (int it = 0; it < nIt; ++it) {
        const int buf = it & 1;
        {   // commit A (fp32 -> f16, fragment layout, 2x ds_store_b128)
            unsigned* dst = aU[buf] + (ar >> 4) * 256;
            uint4 v0 = { packf2(apf0.x, apf0.y), packf2(apf0.z, apf0.w),
                         packf2(apf1.x, apf1.y), packf2(apf1.z, apf1.w) };
            uint4 v1 = { packf2(apf2.x, apf2.y), packf2(apf2.z, apf2.w),
                         packf2(apf3.x, apf3.y), packf2(apf3.z, apf3.w) };
            *(uint4*)(dst + frag_off_u32(ar, ag0))     = v0;
            *(uint4*)(dst + frag_off_u32(ar, ag0 + 1)) = v1;
        }
        {   // commit B (fp4 decode + scale, 2x ds_store_b128)
            const _Float16 sh = (_Float16)(spf * gs * 16384.0f);
            unsigned* dst = bU[buf][bhalf] + (bcc >> 4) * 256;
            uint4 v0 = { mul2h(fp4_decode14(bpf[0]), sh), mul2h(fp4_decode14(bpf[1]), sh),
                         mul2h(fp4_decode14(bpf[2]), sh), mul2h(fp4_decode14(bpf[3]), sh) };
            uint4 v1 = { mul2h(fp4_decode14(bpf[4]), sh), mul2h(fp4_decode14(bpf[5]), sh),
                         mul2h(fp4_decode14(bpf[6]), sh), mul2h(fp4_decode14(bpf[7]), sh) };
            *(uint4*)(dst + frag_off_u32(bcc, bg0))     = v0;
            *(uint4*)(dst + frag_off_u32(bcc, bg0 + 1)) = v1;
        }
        __syncthreads();
        if (it + 1 < nIt) prefetch((it + 1) * 32);   // overlaps wmma below

        const _Float16* aH = (const _Float16*)aU[buf];
        v16h a0 = *(const v16h*)(aH + (2 * mrow + 0) * 512 + lane * 16);
        v16h a1 = *(const v16h*)(aH + (2 * mrow + 1) * 512 + lane * 16);
#pragma unroll
        for (int h = 0; h < 2; ++h) {
            const _Float16* bH = (const _Float16*)bU[buf][h];
#pragma unroll
            for (int nf = 0; nf < 2; ++nf) {
                v16h b = *(const v16h*)(bH + (nsel * 2 + nf) * 512 + lane * 16);
                acc[h][0][nf] = __builtin_amdgcn_wmma_f32_16x16x32_f16(
                    false, a0, false, b, (short)0, acc[h][0][nf], false, false);
                acc[h][1][nf] = __builtin_amdgcn_wmma_f32_16x16x32_f16(
                    false, a1, false, b, (short)0, acc[h][1][nf], false, false);
            }
        }
    }

    // epilogue: inter = up * silu(gate)   (fast rcp, no IEEE div)
    const int hi = lane >> 4, lcol = lane & 15;
#pragma unroll
    for (int mm = 0; mm < 2; ++mm)
#pragma unroll
        for (int nf = 0; nf < 2; ++nf) {
            const int icol = n0 + nsel * 32 + nf * 16 + lcol;
#pragma unroll
            for (int r = 0; r < 8; ++r) {
                const int m = m0 + (2 * mrow + mm) * 16 + r + 8 * hi;
                float g = acc[0][mm][nf][r];
                float u = acc[1][mm][nf][r];
                float sg = g * __builtin_amdgcn_rcpf(1.0f + __expf(-g));
                inter[(size_t)m * I_DIM + icol] = (_Float16)(u * sg);
            }
        }
}

// ---------------------------------------------------------------------------
// GEMM2: out = inter @ dequant(Wd).  Tile 128(M) x 128(N), waves 4x2:
// wave -> mfrags {2m,2m+1}, 4 nfrags -> 8 wmma from 2 A + 4 B loads.
// ---------------------------------------------------------------------------
__global__ __launch_bounds__(256)
void gemm2_kernel(const _Float16* __restrict__ interA, // [E,T,I]
                  const int*      __restrict__ wpkA,   // [E,I/2,H]
                  const float*    __restrict__ wscA,   // [E,I/16,H]
                  const float*    __restrict__ gsp,
                  float*          __restrict__ outA)   // [E,T,H]
{
    const int e = blockIdx.z;
    const _Float16* inter = interA + (size_t)e * T_DIM * I_DIM;
    const int*      wpk   = wpkA   + (size_t)e * (I_DIM / 2) * H_DIM;
    const float*    wsc   = wscA   + (size_t)e * (I_DIM / 16) * H_DIM;
    float*          out   = outA   + (size_t)e * T_DIM * H_DIM;

    const int tid = threadIdx.x, lane = tid & 31, wid = tid >> 5;

    __shared__ __align__(32) unsigned aU[2][8 * 256];   // 128x32 halves x2
    __shared__ __align__(32) unsigned bU[2][8 * 256];   // 32x128 halves x2

    const float gs = gsp[0];
    const int m0 = blockIdx.y * 128;
    const int n0 = blockIdx.x * 128;

    // A writer: f16 source already -> raw uint4 copy into fragment layout
    const int ar  = tid >> 1;
    const int ag0 = (tid & 1) * 2;
    const _Float16* aP = inter + (size_t)(m0 + ar) * I_DIM + 16 * (tid & 1);

    // B writer
    const int bc   = tid & 127;
    const int bsel = tid >> 7;
    const int bg0  = bsel * 2;
    const int ncol = n0 + bc;

    uint4    apf0, apf1;
    unsigned bpf[8];
    float    spf;

    auto prefetch = [&](int k0) {
        const _Float16* ap = aP + k0;
        apf0 = *(const uint4*)(ap + 0);
        apf1 = *(const uint4*)(ap + 8);
        const int kp0 = (k0 >> 1) + 8 * bsel;
#pragma unroll
        for (int u = 0; u < 8; ++u)
            bpf[u] = (unsigned)wpk[(size_t)(kp0 + u) * H_DIM + ncol];
        spf = wsc[(size_t)((k0 >> 4) + bsel) * H_DIM + ncol];
    };

    const int mrow = wid >> 1, nsel = wid & 1;

    v8f acc[2][4];
#pragma unroll
    for (int mm = 0; mm < 2; ++mm)
#pragma unroll
        for (int nf = 0; nf < 4; ++nf) acc[mm][nf] = v8f{};

    prefetch(0);
    const int nIt = I_DIM / 32;
    for (int it = 0; it < nIt; ++it) {
        const int buf = it & 1;
        {   // commit A (pure copy: uint4 -> ds_store_b128)
            unsigned* dst = aU[buf] + (ar >> 4) * 256;
            *(uint4*)(dst + frag_off_u32(ar, ag0))     = apf0;
            *(uint4*)(dst + frag_off_u32(ar, ag0 + 1)) = apf1;
        }
        {   // commit B
            const _Float16 sh = (_Float16)(spf * gs * 16384.0f);
            unsigned* dst = bU[buf] + (bc >> 4) * 256;
            uint4 v0 = { mul2h(fp4_decode14(bpf[0]), sh), mul2h(fp4_decode14(bpf[1]), sh),
                         mul2h(fp4_decode14(bpf[2]), sh), mul2h(fp4_decode14(bpf[3]), sh) };
            uint4 v1 = { mul2h(fp4_decode14(bpf[4]), sh), mul2h(fp4_decode14(bpf[5]), sh),
                         mul2h(fp4_decode14(bpf[6]), sh), mul2h(fp4_decode14(bpf[7]), sh) };
            *(uint4*)(dst + frag_off_u32(bc, bg0))     = v0;
            *(uint4*)(dst + frag_off_u32(bc, bg0 + 1)) = v1;
        }
        __syncthreads();
        if (it + 1 < nIt) prefetch((it + 1) * 32);

        const _Float16* aH = (const _Float16*)aU[buf];
        const _Float16* bH = (const _Float16*)bU[buf];
        v16h a0 = *(const v16h*)(aH + (2 * mrow + 0) * 512 + lane * 16);
        v16h a1 = *(const v16h*)(aH + (2 * mrow + 1) * 512 + lane * 16);
#pragma unroll
        for (int nf = 0; nf < 4; ++nf) {
            v16h b = *(const v16h*)(bH + (nsel * 4 + nf) * 512 + lane * 16);
            acc[0][nf] = __builtin_amdgcn_wmma_f32_16x16x32_f16(
                false, a0, false, b, (short)0, acc[0][nf], false, false);
            acc[1][nf] = __builtin_amdgcn_wmma_f32_16x16x32_f16(
                false, a1, false, b, (short)0, acc[1][nf], false, false);
        }
    }

    const int hi = lane >> 4, lcol = lane & 15;
#pragma unroll
    for (int mm = 0; mm < 2; ++mm)
#pragma unroll
        for (int nf = 0; nf < 4; ++nf) {
            const int n = n0 + nsel * 64 + nf * 16 + lcol;
#pragma unroll
            for (int r = 0; r < 8; ++r) {
                const int m = m0 + (2 * mrow + mm) * 16 + r + 8 * hi;
                out[(size_t)m * H_DIM + n] = acc[mm][nf][r];
            }
        }
}

// ---------------------------------------------------------------------------
// Host launch.  Batched over all experts (grid.z) when the workspace can hold
// inter for all 8 experts (64 MB); otherwise per-expert loop (8 MB).
// ---------------------------------------------------------------------------
extern "C" void kernel_launch(void* const* d_in, const int* in_sizes, int n_in,
                              void* d_out, int out_size, void* d_ws, size_t ws_size,
                              hipStream_t stream) {
    const float* x   = (const float*)d_in[0];
    const int*   gup = (const int*)  d_in[1];
    const float* gus = (const float*)d_in[2];
    const float* ggs = (const float*)d_in[3];
    const int*   dpk = (const int*)  d_in[4];
    const float* dsc = (const float*)d_in[5];
    const float* dgs = (const float*)d_in[6];
    float*       out = (float*)d_out;
    _Float16*    inter = (_Float16*)d_ws;

    (void)in_sizes; (void)n_in; (void)out_size;

    const size_t fullWs = (size_t)E_NUM * T_DIM * I_DIM * sizeof(_Float16);
    if (ws_size >= fullWs) {
        gemm1_silu_kernel<<<dim3(I_DIM / 64, T_DIM / 128, E_NUM), 256, 0, stream>>>(
            x, gup, gus, ggs, inter);
        gemm2_kernel<<<dim3(H_DIM / 128, T_DIM / 128, E_NUM), 256, 0, stream>>>(
            inter, dpk, dsc, dgs, out);
    } else {
        for (int e = 0; e < E_NUM; ++e) {
            const float* xe   = x   + (size_t)e * T_DIM * H_DIM;
            const int*   gupe = gup + (size_t)e * (H_DIM / 2) * (2 * I_DIM);
            const float* guse = gus + (size_t)e * (H_DIM / 16) * (2 * I_DIM);
            const int*   dpe  = dpk + (size_t)e * (I_DIM / 2) * H_DIM;
            const float* dse  = dsc + (size_t)e * (I_DIM / 16) * H_DIM;
            float*       oute = out + (size_t)e * T_DIM * H_DIM;
            gemm1_silu_kernel<<<dim3(I_DIM / 64, T_DIM / 128, 1), 256, 0, stream>>>(
                xe, gupe, guse, ggs, inter);
            gemm2_kernel<<<dim3(H_DIM / 128, T_DIM / 128, 1), 256, 0, stream>>>(
                inter, dpe, dse, dgs, oute);
        }
    }
}
